// EOPA_16655883174581
// MI455X (gfx1250) — matrix operand order repaired and checked by reading, not compiled, not run on hardware
//
#include <hip/hip_runtime.h>
#include <hip/hip_bf16.h>

#define N_NODES 50000
#define DIM     128
#define DEG     32
#define EPSILON 1e-5f

typedef __attribute__((ext_vector_type(16))) _Float16     v16h;
typedef __attribute__((ext_vector_type(8)))  float        v8f;
typedef __attribute__((ext_vector_type(4)))  unsigned int v4u;
typedef __attribute__((ext_vector_type(8)))  int          v8i;
typedef __attribute__((ext_vector_type(4)))  int          v4i;

union HalfPack { v16h h; int4 q[2]; };

// Build a 16x32 f16 A-operand per the CDNA5 ISA layout:
// lane (g = lane/16, m = lane%16) holds row M=m, halves K = {8g..8g+7} U {16+8g..16+8g+7}
// within the 32-wide K chunk c. base points at the start of a 128-half (256B) row.
__device__ __forceinline__ v16h load_a16(const char* base, int c, int g) {
  HalfPack u;
  const char* p = base + c * 64 + g * 16;
  u.q[0] = *(const int4*)(p);
  u.q[1] = *(const int4*)(p + 32);
  return u.h;
}

// B operand: pre-packed so each lane's 16 halves are contiguous (32B).
__device__ __forceinline__ v16h load_b16(const _Float16* w, int tile, int lane) {
  HalfPack u;
  const int4* p = (const int4*)(w + tile * 512 + lane * 16);
  u.q[0] = p[0];
  u.q[1] = p[1];
  return u.h;
}

__device__ __forceinline__ float sigmoidf_(float x) { return 1.0f / (1.0f + __expf(-x)); }

__device__ __forceinline__ v8f wmma_f16(v16h a, v16h b, v8f c) {
  return __builtin_amdgcn_wmma_f32_16x16x32_f16(false, a, false, b, (short)0, c, false, false);
}

// ---------------- stats / normalize ----------------

__global__ void EOPA_zero_stats(float* stats) { stats[threadIdx.x] = 0.0f; }

__global__ void EOPA_col_stats(const float* __restrict__ feat,
                               float* __restrict__ sum, float* __restrict__ sumsq) {
  int c = threadIdx.x;                 // 128 threads = one column each
  int r0 = blockIdx.x * 256;
  int r1 = min(r0 + 256, N_NODES);
  float s = 0.0f, s2 = 0.0f;
  for (int r = r0; r < r1; ++r) {
    float v = feat[(size_t)r * DIM + c];
    s += v; s2 += v * v;
  }
  atomicAdd(&sum[c], s);
  atomicAdd(&sumsq[c], s2);
}

__global__ void EOPA_normalize(const float* __restrict__ feat, const float* __restrict__ sum,
                               const float* __restrict__ sumsq, const float* __restrict__ gamma,
                               const float* __restrict__ beta, _Float16* __restrict__ x16) {
  int i = blockIdx.x * blockDim.x + threadIdx.x;
  if (i >= N_NODES * DIM) return;
  int c = i & (DIM - 1);
  float mean = sum[c] * (1.0f / N_NODES);
  float var  = sumsq[c] * (1.0f / N_NODES) - mean * mean;
  float v = (feat[i] - mean) * rsqrtf(var + EPSILON) * gamma[c] + beta[c];
  x16[i] = (_Float16)v;
}

// ---------------- weight packing into WMMA B layout ----------------
// W is [ncols, 128] (row = output col n, col = k). B[k][n] = W[n][k].
// Tile (nt, kc): lane l holds N = l%16, K_local = 8*(l/16) + (j&7) + 16*(j>>3), j = 0..15,
// stored at dst[tile*512 + l*16 + j].
__global__ void EOPA_pack_b(const float* __restrict__ W, _Float16* __restrict__ dst, int total) {
  int i = blockIdx.x * blockDim.x + threadIdx.x;
  if (i >= total) return;
  int tile = i >> 9, within = i & 511;
  int lane = within >> 4, j = within & 15;
  int nt = tile >> 2, kc = tile & 3;
  int n = nt * 16 + (lane & 15);
  int k = kc * 32 + 8 * (lane >> 4) + (j & 7) + 16 * (j >> 3);
  dst[i] = (_Float16)W[n * DIM + k];
}

// ---------------- fused GRU (32 steps) + output projection ----------------

__global__ __launch_bounds__(256) void EOPA_gru_wmma(
    const _Float16* __restrict__ x16, const int* __restrict__ src,
    const float* __restrict__ b_ih, const float* __restrict__ b_hh,
    const _Float16* __restrict__ wpack,            // Bih (96KB) || Bhh (96KB)
    const _Float16* __restrict__ Bself, const _Float16* __restrict__ Bneigh,
    float* __restrict__ out) {
  extern __shared__ char smem[];
  _Float16* ldsW = (_Float16*)smem;                // 98304 halves (192KB)
  const int tid = threadIdx.x;
  const int wave = tid >> 5, lane = tid & 31;
  const int g = lane >> 4, m = lane & 15;
  _Float16* hbuf = ((_Float16*)(smem + 196608)) + wave * 2048;   // per-wave [16][128] f16

  // Stage both GRU weight matrices into LDS (192KB of the WGP's 320KB) with the
  // Tensor Data Mover: one 2-D descriptor, 24576 x 1 tile of 8-byte elements.
  // TDM ignores EXEC and is tracked per-wave in TENSORcnt, so wave 0 issues+waits,
  // then the block barrier publishes the LDS contents.
  if (tid < 32) {
    unsigned lds_base = (unsigned)(uintptr_t)(void*)smem;        // low 32 bits = LDS byte offset
    unsigned long long ga = (unsigned long long)(uintptr_t)wpack;
    // D# group 0: count=1 | lds_addr | global_addr[56:0] | type=2 ("image")
    v4u g0 = { 1u,
               lds_base,
               (unsigned)ga,
               (unsigned)((ga >> 32) & 0x1FFFFFFu) | (2u << 30) };
    // D# group 1: data_size=3 (8B), tensor_dim0=24576, tensor_dim1=1,
    // tile_dim0=24576, tile_dim1=1, tensor_dim0_stride=24576.
    v8i g1 = { (int)(3u << 16),                      // mask=0 | data_size=3
               (int)((24576u & 0xFFFFu) << 16),      // abar=0 | tensor_dim0[15:0]
               (int)((24576u >> 16) | (1u << 16)),   // tensor_dim0[31:16] | tensor_dim1[15:0]
               (int)(24576u << 16),                  // tensor_dim1[31:16] | tile_dim0
               1,                                    // tile_dim1=1 | tile_dim2=0
               24576,                                // tensor_dim0_stride[31:0]
               0, 0 };                               // stride hi, tensor_dim1_stride
    v4i gz4 = { 0, 0, 0, 0 };                        // <=2-D: groups 2/3 unused
    v8i gz8 = { 0, 0, 0, 0, 0, 0, 0, 0 };
    __builtin_amdgcn_tensor_load_to_lds(g0, g1, gz4, gz4, gz8, 0);
    __builtin_amdgcn_s_wait_tensorcnt(0);
  }

  {
    int4 z = make_int4(0, 0, 0, 0);
    int4* hb = (int4*)hbuf;
    for (int i = lane; i < 256; i += 32) hb[i] = z;              // h0 = 0
  }
  __syncthreads();

  const int rowbase = blockIdx.x * 128 + wave * 16;
  int node = rowbase + m;
  if (node >= N_NODES) node = N_NODES - 1;   // clamp; EXEC stays all-1s for WMMA

  const _Float16* Bih = ldsW;                // tiles: (nt*4 + kc), nt 0..23
  const _Float16* Bhh = ldsW + 49152;
  const char* xbase = (const char*)x16;
  const char* hrow  = (const char*)(hbuf + m * 128);

  // Hidden state in f32 C/D layout: hreg[j][r] holds (M = r + 8g, col = j*16 + m).
  v8f hreg[8];
#pragma unroll
  for (int j = 0; j < 8; ++j) hreg[j] = (v8f){};

  // Biases per column tile (gate order r, z, n).
  float brz[8], bzz[8], bni[8], bnh[8];
#pragma unroll
  for (int j = 0; j < 8; ++j) {
    int col = j * 16 + m;
    brz[j] = b_ih[col] + b_hh[col];
    bzz[j] = b_ih[128 + col] + b_hh[128 + col];
    bni[j] = b_ih[256 + col];
    bnh[j] = b_hh[256 + col];
  }

  for (int t = 0; t < DEG; ++t) {
    int s = src[node * DEG + t];
    // Prefetch next step's gather row (256B) into cache while this step's
    // 192 WMMAs run.
    {
      int snext = src[node * DEG + ((t + 1) & (DEG - 1))];
      const char* pf = xbase + (size_t)snext * 256;
      __builtin_prefetch(pf, 0, 1);
      __builtin_prefetch(pf + 128, 0, 1);
    }
    const char* xrow = xbase + (size_t)s * 256;
    v16h ax[4], ah[4];
#pragma unroll
    for (int c = 0; c < 4; ++c) {
      ax[c] = load_a16(xrow, c, g);   // gathered neighbor row (global, L2-hot)
      ah[c] = load_a16(hrow, c, g);   // previous h (LDS)
    }
#pragma unroll
    for (int j = 0; j < 8; ++j) {
      v8f ir = {}, iz = {}, in_ = {}, hr = {}, hz = {}, hn = {};
#pragma unroll
      for (int c = 0; c < 4; ++c) {
        ir  = wmma_f16(ax[c], load_b16(Bih, (j)      * 4 + c, lane), ir);
        iz  = wmma_f16(ax[c], load_b16(Bih, (j + 8)  * 4 + c, lane), iz);
        in_ = wmma_f16(ax[c], load_b16(Bih, (j + 16) * 4 + c, lane), in_);
        hr  = wmma_f16(ah[c], load_b16(Bhh, (j)      * 4 + c, lane), hr);
        hz  = wmma_f16(ah[c], load_b16(Bhh, (j + 8)  * 4 + c, lane), hz);
        hn  = wmma_f16(ah[c], load_b16(Bhh, (j + 16) * 4 + c, lane), hn);
      }
      v8f hnew;
#pragma unroll
      for (int r = 0; r < 8; ++r) {
        float rr = sigmoidf_(ir[r] + hr[r] + brz[j]);
        float zz = sigmoidf_(iz[r] + hz[r] + bzz[j]);
        float nn = tanhf(in_[r] + bni[j] + rr * (hn[r] + bnh[j]));
        hnew[r] = (1.0f - zz) * nn + zz * hreg[j][r];
      }
      hreg[j] = hnew;
      // write back h as f16 for next step's A-operand build
#pragma unroll
      for (int r = 0; r < 8; ++r)
        hbuf[(r + 8 * g) * 128 + j * 16 + m] = (_Float16)hnew[r];
    }
  }

  // rst = x @ w_self.T + h @ w_neigh.T
  const char* xself = xbase + (size_t)node * 256;
  v16h axs[4], ahf[4];
#pragma unroll
  for (int c = 0; c < 4; ++c) {
    axs[c] = load_a16(xself, c, g);
    ahf[c] = load_a16(hrow, c, g);
  }
#pragma unroll
  for (int j = 0; j < 8; ++j) {
    v8f acc = {};
#pragma unroll
    for (int c = 0; c < 4; ++c) {
      acc = wmma_f16(axs[c], load_b16(Bself,  j * 4 + c, lane), acc);
      acc = wmma_f16(ahf[c], load_b16(Bneigh, j * 4 + c, lane), acc);
    }
#pragma unroll
    for (int r = 0; r < 8; ++r) {
      int row = rowbase + r + 8 * g;
      if (row < N_NODES) out[(size_t)row * DIM + j * 16 + m] = acc[r];
    }
  }
}

// ---------------- host launcher ----------------

extern "C" void kernel_launch(void* const* d_in, const int* in_sizes, int n_in,
                              void* d_out, int out_size, void* d_ws, size_t ws_size,
                              hipStream_t stream) {
  const float* feat    = (const float*)d_in[0];
  const int*   src     = (const int*)d_in[1];
  const float* gamma   = (const float*)d_in[2];
  const float* beta    = (const float*)d_in[3];
  const float* w_ih    = (const float*)d_in[4];
  const float* w_hh    = (const float*)d_in[5];
  const float* b_ih    = (const float*)d_in[6];
  const float* b_hh    = (const float*)d_in[7];
  const float* w_self  = (const float*)d_in[8];
  const float* w_neigh = (const float*)d_in[9];
  float* out = (float*)d_out;
  char* ws = (char*)d_ws;

  float* sum    = (float*)(ws + 0);          // 128 f32
  float* sumsq  = (float*)(ws + 512);        // 128 f32
  _Float16* x16 = (_Float16*)(ws + 4096);    // N*128 f16 = 12.8MB
  const size_t OFF_B = 4096 + (size_t)N_NODES * DIM * 2;
  _Float16* Bih    = (_Float16*)(ws + OFF_B);                     // 96KB
  _Float16* Bhh    = (_Float16*)(ws + OFF_B + 98304);             // 96KB
  _Float16* Bself  = (_Float16*)(ws + OFF_B + 2 * 98304);         // 32KB
  _Float16* Bneigh = (_Float16*)(ws + OFF_B + 2 * 98304 + 32768); // 32KB

  EOPA_zero_stats<<<1, 256, 0, stream>>>(sum);
  EOPA_col_stats<<<(N_NODES + 255) / 256, 128, 0, stream>>>(feat, sum, sumsq);
  EOPA_normalize<<<(N_NODES * DIM + 255) / 256, 256, 0, stream>>>(feat, sum, sumsq, gamma, beta, x16);
  EOPA_pack_b<<<(3 * DIM * DIM + 255) / 256, 256, 0, stream>>>(w_ih, Bih, 3 * DIM * DIM);
  EOPA_pack_b<<<(3 * DIM * DIM + 255) / 256, 256, 0, stream>>>(w_hh, Bhh, 3 * DIM * DIM);
  EOPA_pack_b<<<(DIM * DIM + 255) / 256, 256, 0, stream>>>(w_self, Bself, DIM * DIM);
  EOPA_pack_b<<<(DIM * DIM + 255) / 256, 256, 0, stream>>>(w_neigh, Bneigh, DIM * DIM);
  // 8 waves/block, 16 nodes/wave -> 128 nodes/block; 224KB dynamic LDS (weights + h buffers)
  EOPA_gru_wmma<<<(N_NODES + 127) / 128, 256, 229376, stream>>>(
      x16, src, b_ih, b_hh, Bih, Bself, Bneigh, out);
}
